// LIIF_14886356648420
// MI455X (gfx1250) — compile-verified, matrix operand order checked by reading, not run on hardware
//
#include <hip/hip_runtime.h>
#include <hip/hip_bf16.h>

typedef _Float16 v16h __attribute__((ext_vector_type(16)));
typedef _Float16 v8h  __attribute__((ext_vector_type(8)));
typedef float    v8f  __attribute__((ext_vector_type(8)));

#define C_ENC 64
#define HFE 96
#define WFE 96
#define QN 32768
#define BQ 65536          // B * Q
#define NROWS 262144      // 4 * B * Q
#define HID 256
#define IN_DIM 580
#define KPAD 608
#define K0C 19            // KPAD / 32
#define KHC 8             // HID / 32
#define TILE_M 64
#define XS_STRIDE 608
#define HS_STRIDE 272

// workspace layout (bytes)
#define W0P_HALFS (K0C * 16 * 32 * 16)   // 155648 halfs
#define WHP_HALFS (KHC * 16 * 32 * 16)   // 65536 halfs
#define PRED_OFF  704512                 // (W0P + 3*WHP) * 2 bytes
#define AREA_OFF  1753088                // PRED_OFF + NROWS*4

// ---------------------------------------------------------------------------
// Pack W0..W3 (f32, K x 256 row-major) into B-fragment-major f16 records:
// dst[((kc*16 + nt)*32 + lane)*16 + e], where lane<16 holds column nt*16+lane
// with k_local in {0..7,16..23} and lane>=16 holds k_local in {8..15,24..31},
// matching the 16-bit 16x32 WMMA operand layout. K beyond Ksrc is zero-padded.
// ---------------------------------------------------------------------------
__global__ void liif_pack_weights(const float* __restrict__ W0,
                                  const float* __restrict__ W1,
                                  const float* __restrict__ W2,
                                  const float* __restrict__ W3,
                                  _Float16* __restrict__ wsH) {
    int tid = blockIdx.x * blockDim.x + threadIdx.x;
    const int total = (K0C + 3 * KHC) * 16 * 32;
    if (tid >= total) return;
    int lane = tid & 31;
    int nt   = (tid >> 5) & 15;
    int kcA  = tid >> 9;
    const float* src;
    _Float16* dst;
    int kc, Ksrc;
    if (kcA < K0C) {
        src = W0; dst = wsH; kc = kcA; Ksrc = IN_DIM;
    } else {
        int j = kcA - K0C;
        int L = j >> 3;
        kc = j & 7; Ksrc = HID;
        src = (L == 0) ? W1 : (L == 1) ? W2 : W3;
        dst = wsH + W0P_HALFS + (size_t)L * WHP_HALFS;
    }
    int l16 = lane & 15, hgrp = lane >> 4;
    _Float16* out = dst + ((size_t)(kc * 16 + nt) * 32 + lane) * 16;
    int n = nt * 16 + l16;
#pragma unroll
    for (int e = 0; e < 16; ++e) {
        int klocal = hgrp * 8 + (e & 7) + ((e >> 3) << 4);
        int K = kc * 32 + klocal;
        float v = (K < Ksrc) ? src[(size_t)K * HID + n] : 0.f;
        out[e] = (_Float16)v;
    }
}

// ---------------------------------------------------------------------------
// One MLP layer as tiled WMMA GEMM. Wave w owns N-tiles {2w, 2w+1} (32
// columns) and ALL four 16-row M-strips: the 8 waves of the block together
// read the packed weight set exactly ONCE per K-pass (4x less L2 weight
// traffic than a per-wave M-strip split), while A fragments are re-read from
// LDS (cheap). 8 f32 accumulator tiles per wave.
// ---------------------------------------------------------------------------
__device__ __forceinline__ void gemm_layer(const _Float16* __restrict__ srcLDS, int srcStride,
                                           _Float16* __restrict__ dstLDS, int dstStride,
                                           const _Float16* __restrict__ Wp,
                                           const float* __restrict__ bias,
                                           int kChunks, int wv, int lane, bool relu) {
    v8f acc[4][2] = {};
    int rbase = lane & 15;
    int koffBase = (lane >> 4) << 3;
    for (int kc = 0; kc < kChunks; ++kc) {
        int kb = kc * 32 + koffBase;
        v16h a[4];
#pragma unroll
        for (int ms = 0; ms < 4; ++ms) {
            const _Float16* p = srcLDS + (ms * 16 + rbase) * srcStride + kb;
            v8h lo = *(const v8h*)p;
            v8h hi = *(const v8h*)(p + 16);
#pragma unroll
            for (int i = 0; i < 8; ++i) { a[ms][i] = lo[i]; a[ms][8 + i] = hi[i]; }
        }
#pragma unroll
        for (int ntl = 0; ntl < 2; ++ntl) {
            int ntg = wv * 2 + ntl;
            v16h bf = *(const v16h*)(Wp + ((size_t)(kc * 16 + ntg) * 32 + lane) * 16);
#pragma unroll
            for (int ms = 0; ms < 4; ++ms) {
                acc[ms][ntl] = __builtin_amdgcn_wmma_f32_16x16x32_f16(
                    false, a[ms], false, bf, (short)0, acc[ms][ntl], false, false);
            }
        }
    }
    int l16 = lane & 15, mo = (lane >> 4) << 3;
#pragma unroll
    for (int ntl = 0; ntl < 2; ++ntl) {
        int n = (wv * 2 + ntl) * 16 + l16;
        float bv = bias[n];
#pragma unroll
        for (int ms = 0; ms < 4; ++ms) {
#pragma unroll
            for (int r = 0; r < 8; ++r) {
                float v = acc[ms][ntl][r] + bv;
                if (relu) v = fmaxf(v, 0.f);
                dstLDS[(ms * 16 + r + mo) * dstStride + n] = (_Float16)v;
            }
        }
    }
}

// ---------------------------------------------------------------------------
// Fused megakernel: gather ensemble-shifted unfold features into LDS, then run
// the 4 hidden layers with WMMA, then the 256->1 head. One block = 64 rows.
// ---------------------------------------------------------------------------
__global__ void __launch_bounds__(256)
liif_mlp(const float* __restrict__ feat, const float* __restrict__ coord,
         const float* __restrict__ cell, const _Float16* __restrict__ wsH,
         const float* __restrict__ b0, const float* __restrict__ b1,
         const float* __restrict__ b2, const float* __restrict__ b3,
         const float* __restrict__ W4, const float* __restrict__ b4,
         float* __restrict__ preds, float* __restrict__ areas) {
    __shared__ alignas(16) _Float16 Xs[TILE_M * XS_STRIDE];
    __shared__ alignas(16) _Float16 Hs[TILE_M * HS_STRIDE];
    int t = threadIdx.x;

    // ---- stage 1: build input tile (4 threads per row) ----
    {
        int r = t >> 2, part = t & 3;
        int R  = blockIdx.x * TILE_M + r;
        int s  = R >> 16;
        int bq = R & (BQ - 1);
        int b  = bq >> 15;
        int q  = bq & (QN - 1);
        float vx = (s & 2) ? 1.f : -1.f;
        float vy = (s & 1) ? 1.f : -1.f;
        float c0  = coord[((size_t)b * QN + q) * 2 + 0];
        float c1  = coord[((size_t)b * QN + q) * 2 + 1];
        float ce0 = cell[((size_t)b * QN + q) * 2 + 0];
        float ce1 = cell[((size_t)b * QN + q) * 2 + 1];
        const float lo = -1.f + 1e-6f, hi = 1.f - 1e-6f;
        float g0 = fminf(fmaxf(c0 + vx * (1.f / HFE) + 1e-6f, lo), hi);
        float g1 = fminf(fmaxf(c1 + vy * (1.f / WFE) + 1e-6f, lo), hi);
        int iy = min(max((int)rintf(((g0 + 1.f) * HFE - 1.f) * 0.5f), 0), HFE - 1);
        int ix = min(max((int)rintf(((g1 + 1.f) * WFE - 1.f) * 0.5f), 0), WFE - 1);
        float qc0 = ((float)iy + 0.5f) * (2.f / HFE) - 1.f;
        float qc1 = ((float)ix + 0.5f) * (2.f / WFE) - 1.f;
        float rel0 = (c0 - qc0) * (float)HFE;
        float rel1 = (c1 - qc1) * (float)WFE;
        const float* fb = feat + (size_t)b * C_ENC * HFE * WFE;
        for (int c = part * 16; c < part * 16 + 16; ++c) {
            const float* fc = fb + (size_t)c * HFE * WFE;
            int o = r * XS_STRIDE + c * 9;
#pragma unroll
            for (int ky = 0; ky < 3; ++ky) {
                int y = iy + ky - 1;
                bool yok = (y >= 0) && (y < HFE);
#pragma unroll
                for (int kx = 0; kx < 3; ++kx) {
                    int x = ix + kx - 1;
                    float v = (yok && x >= 0 && x < WFE) ? fc[y * WFE + x] : 0.f;
                    Xs[o + ky * 3 + kx] = (_Float16)v;
                }
            }
        }
        if (part == 0) {
            Xs[r * XS_STRIDE + 576] = (_Float16)rel0;
            Xs[r * XS_STRIDE + 577] = (_Float16)rel1;
            Xs[r * XS_STRIDE + 578] = (_Float16)(ce0 * (float)HFE);
            Xs[r * XS_STRIDE + 579] = (_Float16)(ce1 * (float)WFE);
            for (int j = IN_DIM; j < KPAD; ++j) Xs[r * XS_STRIDE + j] = (_Float16)0.f;
            areas[R] = fabsf(rel0 * rel1) + 1e-9f;
        }
    }
    __syncthreads();

    // ---- stages 2-5: WMMA layers, ping-pong Xs <-> Hs ----
    int lane = t & 31, wv = t >> 5;
    const _Float16* W0p = wsH;
    const _Float16* W1p = wsH + W0P_HALFS;
    const _Float16* W2p = W1p + WHP_HALFS;
    const _Float16* W3p = W2p + WHP_HALFS;
    gemm_layer(Xs, XS_STRIDE, Hs, HS_STRIDE, W0p, b0, K0C, wv, lane, true);
    __syncthreads();
    gemm_layer(Hs, HS_STRIDE, Xs, XS_STRIDE, W1p, b1, KHC, wv, lane, true);
    __syncthreads();
    gemm_layer(Xs, XS_STRIDE, Hs, HS_STRIDE, W2p, b2, KHC, wv, lane, true);
    __syncthreads();
    gemm_layer(Hs, HS_STRIDE, Xs, XS_STRIDE, W3p, b3, KHC, wv, lane, true);
    __syncthreads();

    // ---- stage 6: 256 -> 1 head ----
    if (t < TILE_M) {
        float accv = 0.f;
        const _Float16* hrow = Xs + t * XS_STRIDE;
        for (int k = 0; k < HID; ++k) accv += (float)hrow[k] * W4[k];
        preds[blockIdx.x * TILE_M + t] = accv + b4[0];
    }
}

// ---------------------------------------------------------------------------
// Local-ensemble combine with diagonal area swap.
// ---------------------------------------------------------------------------
__global__ void liif_combine(const float* __restrict__ preds,
                             const float* __restrict__ areas,
                             float* __restrict__ out) {
    int i = blockIdx.x * blockDim.x + threadIdx.x;
    if (i >= BQ) return;
    float a0 = areas[i], a1 = areas[BQ + i], a2 = areas[2 * BQ + i], a3 = areas[3 * BQ + i];
    float p0 = preds[i], p1 = preds[BQ + i], p2 = preds[2 * BQ + i], p3 = preds[3 * BQ + i];
    float tot = a0 + a1 + a2 + a3;
    out[i] = (p0 * a3 + p1 * a2 + p2 * a1 + p3 * a0) / tot;
}

extern "C" void kernel_launch(void* const* d_in, const int* in_sizes, int n_in,
                              void* d_out, int out_size, void* d_ws, size_t ws_size,
                              hipStream_t stream) {
    const float* feat  = (const float*)d_in[0];
    const float* coord = (const float*)d_in[1];
    const float* cell  = (const float*)d_in[2];
    const float* W0 = (const float*)d_in[3];
    const float* b0 = (const float*)d_in[4];
    const float* W1 = (const float*)d_in[5];
    const float* b1 = (const float*)d_in[6];
    const float* W2 = (const float*)d_in[7];
    const float* b2 = (const float*)d_in[8];
    const float* W3 = (const float*)d_in[9];
    const float* b3 = (const float*)d_in[10];
    const float* W4 = (const float*)d_in[11];
    const float* b4 = (const float*)d_in[12];

    _Float16* wsH  = (_Float16*)d_ws;
    float* preds   = (float*)((char*)d_ws + PRED_OFF);
    float* areas   = (float*)((char*)d_ws + AREA_OFF);
    float* out     = (float*)d_out;

    // 1) pack weights into WMMA fragment layout (f16)
    const int packThreads = (K0C + 3 * KHC) * 16 * 32;   // 22016
    liif_pack_weights<<<(packThreads + 255) / 256, 256, 0, stream>>>(W0, W1, W2, W3, wsH);

    // 2) fused gather + 4 WMMA layers + head, one block per 64 rows
    liif_mlp<<<NROWS / TILE_M, 256, 0, stream>>>(feat, coord, cell, wsH,
                                                 b0, b1, b2, b3, W4, b4, preds, areas);

    // 3) ensemble combine
    liif_combine<<<BQ / 256, 256, 0, stream>>>(preds, areas, out);
}
